// CAMLocalBiHead_14018773254294
// MI455X (gfx1250) — compile-verified, matrix operand order checked by reading, not run on hardware
//
#include <hip/hip_runtime.h>
#include <hip/hip_bf16.h>
#include <math.h>

// ---------------------------------------------------------------------------
// CAMLocalBiHead loss for MI455X (gfx1250), wave32 + WMMA bf16.
// Dominant work: 3x3 conv (C=256 -> D=512) as implicit GEMM
//   M=100352 (B*T*H*W), N=512, K=2304 (C*9)  => 237 GFLOP
// v_wmma_f32_16x16x32_bf16 with f32 accumulation. All 8 B fragments of a
// K-chunk are loaded into distinct registers before the 8 WMMAs so the load
// clause completes once per chunk and the WMMAs issue back-to-back.
// ---------------------------------------------------------------------------

typedef __attribute__((ext_vector_type(16))) __bf16 v16bf;
typedef __attribute__((ext_vector_type(8)))  float  v8f;

#define Bn    64
#define Cc    256
#define Tt    32
#define Hh    7
#define Ww    7
#define NSPAT (Tt*Hh*Ww)        // 1568
#define NPOS  (Bn*NSPAT)        // 100352
#define Dd    512
#define KK    (Cc*9)            // 2304
#define KV    97
#define KN    300
#define NTOK  392
#define MTILE 32

union Frag { v16bf v; unsigned int u[8]; };

// round-to-nearest-even f32 -> bf16 bits
__device__ __forceinline__ unsigned short f2bf(float f) {
  unsigned int u = __float_as_uint(f);
  unsigned int r = u + 0x7fffu + ((u >> 16) & 1u);
  if ((u & 0x7f800000u) == 0x7f800000u) r = u;   // inf / nan passthrough
  return (unsigned short)(r >> 16);
}

// Per-lane 16-bit fragment gather (ISA 7.12.2 K order {0..7,16..23}/{8..15,24..31});
// each half is 16B-contiguous so these merge into two global_load_b128.
__device__ __forceinline__ void load_frag(Frag& f, const unsigned short* base,
                                          int hi8) {
  #pragma unroll
  for (int v = 0; v < 8; ++v) {
    const int keven = ((v >> 2) << 4) + hi8 + ((v & 3) << 1);
    f.u[v] = *(const unsigned int*)(base + keven);
  }
}

// Branchless padded A load: always load from a safe address, mask to zero.
__device__ __forceinline__ void load_frag_masked(Frag& f, const unsigned short* base,
                                                 int hi8, unsigned int mask) {
  #pragma unroll
  for (int v = 0; v < 8; ++v) {
    const int keven = ((v >> 2) << 4) + hi8 + ((v & 3) << 1);
    f.u[v] = (*(const unsigned int*)(base + keven)) & mask;
  }
}

// A-matrix source pointer for K-chunk kc (chunk never straddles a (dh,dw) tap).
// Returns a clamped-safe pointer; mask = ~0u when in-bounds else 0.
__device__ __forceinline__ const unsigned short* a_ptr(
    const unsigned short* xh, int mrow, int hpos, int wpos, int kc,
    unsigned int& mask) {
  const int kbase = kc * 32;
  const int dhw   = kbase >> 8;
  const int coff  = kbase & 255;
  const int dh    = dhw / 3 - 1;
  const int dw    = dhw % 3 - 1;
  const int hh    = hpos + dh, ww = wpos + dw;
  const bool valid = (hh >= 0) & (hh < Hh) & (ww >= 0) & (ww < Ww);
  mask = valid ? 0xffffffffu : 0u;
  const int npos = valid ? (mrow + dh * Ww + dw) : mrow;  // always in-bounds
  return xh + (size_t)npos * Cc + coff;
}

// x (B,C,T,H,W) f32 -> xh[pos][c] bf16, channels-last
__global__ __launch_bounds__(256) void prep_x(const float* __restrict__ x,
                                              unsigned short* __restrict__ xh) {
  const int pos = blockIdx.x;           // ((b*T+t)*H+h)*W+w
  const int c   = threadIdx.x;
  const int b   = pos / NSPAT;
  const int thw = pos % NSPAT;
  const float v = x[((size_t)(b * Cc + c)) * NSPAT + thw];
  xh[(size_t)pos * Cc + c] = f2bf(v);
}

// conv_w (D,C,1,3,3) f32 -> wkT[d][k] bf16,  k = (dh*3+dw)*256 + c
__global__ __launch_bounds__(256) void prep_w(const float* __restrict__ cw,
                                              unsigned short* __restrict__ wkT) {
  const int i = blockIdx.x * 256 + threadIdx.x;
  if (i >= Dd * KK) return;
  const int d   = i / KK;
  const int k   = i % KK;
  const int dhw = k >> 8;       // 0..8
  const int c   = k & 255;
  wkT[i] = f2bf(cw[((size_t)(d * Cc + c)) * 9 + dhw]);
}

// cam_gt = max(cam_v, cam_n); one block per batch sample.
__global__ __launch_bounds__(256) void cam_kernel(const float* __restrict__ x,
                                                  const float* __restrict__ lv,
                                                  const float* __restrict__ ln,
                                                  const float* __restrict__ wv,
                                                  const float* __restrict__ wn,
                                                  float* __restrict__ cam) {
  __shared__ float s_w[Cc];
  __shared__ float s_row[NSPAT];
  __shared__ float s_cam[NSPAT];
  __shared__ float s_red[256];
  __shared__ int   s_idx[256];
  __shared__ float s_mm[2];
  const int b   = blockIdx.x;
  const int tid = threadIdx.x;

  for (int head = 0; head < 2; ++head) {
    const int K = head ? KN : KV;
    const float* lg = head ? (ln + (size_t)b * KN) : (lv + (size_t)b * KV);
    const float* wp = head ? wn : wv;

    // ---- argmax over class logits (tie-break: lower index, as jnp.argmax) ----
    float bv = -INFINITY; int bi = 0x7fffffff;
    for (int k = tid; k < K; k += 256) {
      float v = lg[k];
      if (v > bv || (v == bv && k < bi)) { bv = v; bi = k; }
    }
    s_red[tid] = bv; s_idx[tid] = bi;
    __syncthreads();
    for (int s = 128; s > 0; s >>= 1) {
      if (tid < s) {
        float ov = s_red[tid + s]; int oi = s_idx[tid + s];
        if (ov > s_red[tid] || (ov == s_red[tid] && oi < s_idx[tid])) {
          s_red[tid] = ov; s_idx[tid] = oi;
        }
      }
      __syncthreads();
    }
    const int top = s_idx[0];
    __syncthreads();
    if (tid < Cc) s_w[tid] = wp[(size_t)top * Cc + tid];
    __syncthreads();

    // ---- row[n] = sum_c w_top[c] * x[b,c,n] ----
    float mn = INFINITY, mx = -INFINITY;
    for (int n = tid; n < NSPAT; n += 256) {
      const float* xb = x + (size_t)b * Cc * NSPAT + n;
      float acc = 0.f;
      #pragma unroll 4
      for (int c = 0; c < Cc; ++c) acc += s_w[c] * xb[(size_t)c * NSPAT];
      s_row[n] = acc;
      mn = fminf(mn, acc); mx = fmaxf(mx, acc);
    }
    s_red[tid] = mn; __syncthreads();
    for (int s = 128; s > 0; s >>= 1) { if (tid < s) s_red[tid] = fminf(s_red[tid], s_red[tid + s]); __syncthreads(); }
    if (tid == 0) s_mm[0] = s_red[0];
    __syncthreads();
    s_red[tid] = mx; __syncthreads();
    for (int s = 128; s > 0; s >>= 1) { if (tid < s) s_red[tid] = fmaxf(s_red[tid], s_red[tid + s]); __syncthreads(); }
    if (tid == 0) s_mm[1] = s_red[0];
    __syncthreads();
    const float rmn = s_mm[0], rmx = s_mm[1];
    const float inv = 1.0f / (rmx - rmn);

    // ---- stable top-392 selection by rank (== top_k + scatter) ----
    for (int n = tid; n < NSPAT; n += 256) {
      const float v = s_row[n];
      int cnt = 0;
      for (int j = 0; j < NSPAT; ++j) {
        const float w = s_row[j];
        cnt += (w > v) || (w == v && j < n);
      }
      const float val = (cnt < NTOK) ? (v - rmn) * inv : 0.f;
      if (head == 0) s_cam[n] = val;
      else           s_cam[n] = fmaxf(s_cam[n], val);
    }
    __syncthreads();
  }
  for (int n = tid; n < NSPAT; n += 256)
    cam[(size_t)b * NSPAT + n] = s_cam[n];
}

// Fused implicit-GEMM conv3x3 + bias + ReLU + score-dot + BCE partial sum.
// Block: 256 threads (8 waves, wave grid 2M x 4N), M-tile = 32 positions.
__global__ __launch_bounds__(256) void conv_bce_kernel(
    const unsigned short* __restrict__ xh, const unsigned short* __restrict__ wkT,
    const float* __restrict__ convb, const float* __restrict__ scorew,
    const float* __restrict__ scoreb, const float* __restrict__ cam,
    double* __restrict__ lossAcc) {
  __shared__ float s_logit[MTILE];
  const int tid  = threadIdx.x;
  const int lane = tid & 31;
  const int wave = tid >> 5;
  const int wm   = wave >> 2;               // 0..1  (M sub-tile)
  const int wn   = wave & 3;                // 0..3  (N 128-col slice)
  const int l16  = lane & 15;
  const int hi8  = (lane >> 4) << 3;        // 0 or 8 (K-half select)
  const int m0   = blockIdx.x * MTILE;
  const int mrow = m0 + wm * 16 + l16;      // output position (A row)
  const int wpos = mrow % Ww;
  const int hpos = (mrow / Ww) % Hh;
  const unsigned short* wbase = wkT + (size_t)(wn * 128 + l16) * KK;

  if (tid < MTILE) s_logit[tid] = 0.f;
  __syncthreads();

  v8f acc[8] = {};
  Frag aa[2];
  Frag b[8];

  {
    unsigned int m;
    const unsigned short* ab = a_ptr(xh, mrow, hpos, wpos, 0, m);
    load_frag_masked(aa[0], ab, hi8, m);
  }

  #pragma unroll 2
  for (int kc = 0; kc < KK / 32; ++kc) {
    const int kbase = kc * 32;
    // Load ALL 8 B fragments for this chunk into distinct registers:
    // one clause of 16 global_load_b128, one wait, then 8 WMMAs back-to-back.
    #pragma unroll
    for (int nt = 0; nt < 8; ++nt)
      load_frag(b[nt], wbase + (size_t)nt * 16 * KK + kbase, hi8);
    // Prefetch A fragment for the next K-chunk into the other buffer.
    if (kc + 1 < KK / 32) {
      unsigned int m;
      const unsigned short* ab = a_ptr(xh, mrow, hpos, wpos, kc + 1, m);
      load_frag_masked(aa[(kc + 1) & 1], ab, hi8, m);
    }
    #pragma unroll
    for (int nt = 0; nt < 8; ++nt)
      acc[nt] = __builtin_amdgcn_wmma_f32_16x16x32_bf16(
          false, aa[kc & 1].v, false, b[nt].v, (short)0, acc[nt], false, false);
  }

  // Epilogue: h1 = relu(acc + conv_b[d]);  logit += h1 * score_w[d]
  // C/D layout: lane n = l16, VGPR r -> row (hi8 + r).
  float part[8] = {0, 0, 0, 0, 0, 0, 0, 0};
  #pragma unroll
  for (int nt = 0; nt < 8; ++nt) {
    const int d  = wn * 128 + nt * 16 + l16;
    const float cb = convb[d];
    const float sw = scorew[d];
    #pragma unroll
    for (int r = 0; r < 8; ++r) {
      float h1 = acc[nt][r] + cb;
      h1 = h1 > 0.f ? h1 : 0.f;
      part[r] += h1 * sw;
    }
  }
  #pragma unroll
  for (int r = 0; r < 8; ++r)
    atomicAdd(&s_logit[wm * 16 + hi8 + r], part[r]);
  __syncthreads();

  if (tid < MTILE) {
    const int pos  = m0 + tid;
    const float xv = s_logit[tid] + scoreb[0];
    const float y  = cam[pos];
    const float ls_pos = fminf(xv, 0.f) - log1pf(expf(-fabsf(xv)));  // log_sigmoid(x)
    const float ls_neg = fminf(-xv, 0.f) - log1pf(expf(-fabsf(xv))); // log_sigmoid(-x)
    const float bce = -(y * ls_pos + (1.f - y) * ls_neg);
    atomicAdd(lossAcc, (double)bce);
  }
}

__global__ void zero_acc(double* a) { *a = 0.0; }

__global__ void finalize_kernel(const double* __restrict__ a, float* __restrict__ out) {
  out[0] = (float)(*a / (double)NPOS);   // LOSS_WEIGHT == 1.0
}

extern "C" void kernel_launch(void* const* d_in, const int* in_sizes, int n_in,
                              void* d_out, int out_size, void* d_ws, size_t ws_size,
                              hipStream_t stream) {
  const float* x   = (const float*)d_in[0];
  const float* plv = (const float*)d_in[1];
  const float* pln = (const float*)d_in[2];
  const float* wv  = (const float*)d_in[3];
  const float* wn  = (const float*)d_in[4];
  const float* cw  = (const float*)d_in[5];
  const float* cb  = (const float*)d_in[6];
  const float* sw  = (const float*)d_in[7];
  const float* sb  = (const float*)d_in[8];
  float* out = (float*)d_out;

  // Workspace layout (~54.2 MB total)
  char* ws = (char*)d_ws;
  const size_t XH_BYTES  = (size_t)NPOS * Cc * 2;   // 51,380,224
  const size_t WK_BYTES  = (size_t)Dd * KK * 2;     //  2,359,296
  const size_t CAM_BYTES = (size_t)NPOS * 4;        //    401,408
  unsigned short* xh   = (unsigned short*)ws;
  unsigned short* wkT  = (unsigned short*)(ws + XH_BYTES);
  float*          camb = (float*)(ws + XH_BYTES + WK_BYTES);
  double*         acc  = (double*)(ws + XH_BYTES + WK_BYTES + CAM_BYTES);

  zero_acc<<<1, 1, 0, stream>>>(acc);
  prep_x<<<NPOS, 256, 0, stream>>>(x, xh);
  prep_w<<<(Dd * KK + 255) / 256, 256, 0, stream>>>(cw, wkT);
  cam_kernel<<<Bn, 256, 0, stream>>>(x, plv, pln, wv, wn, camb);
  conv_bce_kernel<<<NPOS / MTILE, 256, 0, stream>>>(xh, wkT, cb, sw, sb, camb, acc);
  finalize_kernel<<<1, 1, 0, stream>>>(acc, out);
}
